// NCEAverageFullSoftmax_20864951124342
// MI455X (gfx1250) — matrix operand order repaired and checked by reading, not compiled, not run on hardware
//
#include <hip/hip_runtime.h>
#include <hip/hip_bf16.h>
#include <math.h>

// Problem constants (match reference)
#define BN   256        // batch
#define DN   128        // feature dim
#define KN   400000     // memory bank rows
#define INV_T (1.0f / 0.07f)

// Tiling
#define KCHUNK 128      // memory rows per block  (400000 / 128 = 3125 blocks)
#define NKT    (KCHUNK / 16)   // 8 k-tiles per block
#define ABLOB  (16 * 4 * 32 * 16)  // bf16 elements per packed activation (64 KB)

typedef __attribute__((ext_vector_type(16))) __bf16 v16bf;
typedef __attribute__((ext_vector_type(8)))  float  v8f;
typedef __attribute__((ext_vector_type(4)))  __bf16 bf16x4;
typedef __attribute__((ext_vector_type(4)))  float  f32x4;   // native ext-vector (NT-capable)

// ---------------------------------------------------------------------------
// Kernel 1: pack l / ab (f32 256x128) into the CDNA5 16-bit A-fragment layout
// (16x32 per fragment; lane = m + 16*((kc>>3)&1), slot = (kc&7) + 8*(kc>>4))
// stored as bf16 in workspace. Tiny: 2 blocks x 256 threads.
// ---------------------------------------------------------------------------
__global__ __launch_bounds__(256)
void pack_feat_kernel(const float* __restrict__ l,
                      const float* __restrict__ ab,
                      __bf16* __restrict__ wsA) {
    const float* feat = (blockIdx.x == 0) ? l : ab;
    __bf16* dst = wsA + (size_t)blockIdx.x * ABLOB;
    const int b  = threadIdx.x;
    const int bt = b >> 4;       // b-tile
    const int m  = b & 15;       // row within tile
#pragma unroll 8
    for (int d = 0; d < DN; ++d) {
        float v = feat[(size_t)b * DN + d];
        int c    = d >> 5;       // which K=32 chunk
        int kc   = d & 31;
        int lane = m + (((kc >> 3) & 1) << 4);
        int slot = (kc & 7) + ((kc >> 4) << 3);
        dst[((size_t)((bt * 4 + c) * 32 + lane) << 4) + slot] = (__bf16)v;
    }
}

// ---------------------------------------------------------------------------
// Kernel 2: fused  out[p] = feat[p] @ mem[p]^T / T   +   new_memory copy.
// grid = (3125, 2); block = 256 (8 waves).
//   p==0: feat=l  (ws blob 0), mem=memory_ab -> out_l  @0,     copy->new_memory_ab
//   p==1: feat=ab (ws blob 1), mem=memory_l  -> out_ab @B*K,   copy->new_memory_l
// Each block stages 128 memory rows: f32 copy streamed out (NT) + bf16
// B-fragment swizzle into LDS (lane = n + 16*((kc>>3)&1), slot = (kc&7) + 8*(kc>>4)).
// ---------------------------------------------------------------------------
__global__ __launch_bounds__(256)
void gemm_copy_kernel(const __bf16* __restrict__ wsA,
                      const float*  __restrict__ mem_p0,   // memory_ab
                      const float*  __restrict__ mem_p1,   // memory_l
                      float*        __restrict__ out) {
    __shared__ alignas(32) __bf16 Bs[NKT][4][32][16];   // 32 KB

    const int p = blockIdx.y;
    const float* mem = p ? mem_p1 : mem_p0;
    float* outp      = out + (size_t)p * ((size_t)BN * KN);
    float* copy_dst  = out + 2ull * BN * KN + (p ? 0ull : (size_t)KN * DN);
    const __bf16* aBase = wsA + (size_t)p * ABLOB;

    const int tid = threadIdx.x;
    const int k0  = blockIdx.x * KCHUNK;

    // ---- stage: one HBM pass over this 128x128 f32 chunk ----
    const f32x4* src4 = (const f32x4*)(mem + (size_t)k0 * DN);
    f32x4*       dst4 = (f32x4*)(copy_dst + (size_t)k0 * DN);
#pragma unroll
    for (int i = 0; i < (KCHUNK * DN / 4) / 256; ++i) {
        int e4 = tid + i * 256;
        f32x4 v = __builtin_nontemporal_load(&src4[e4]);
        __builtin_nontemporal_store(v, &dst4[e4]);
        int e  = e4 << 2;
        int r  = e >> 7;          // row within chunk
        int d  = e & (DN - 1);
        int kt = r >> 4, n = r & 15;
        int c  = d >> 5, kc = d & 31;      // kc aligned to 4
        int lane = n + (((kc >> 3) & 1) << 4);
        int slot = (kc & 7) + ((kc >> 4) << 3); // slot..slot+3, same lane
        bf16x4 bv;
        bv[0] = (__bf16)v[0]; bv[1] = (__bf16)v[1];
        bv[2] = (__bf16)v[2]; bv[3] = (__bf16)v[3];
        *(bf16x4*)&Bs[kt][c][lane][slot] = bv;
    }
    __syncthreads();

    // ---- compute: wave w owns b-tiles {2w, 2w+1}, sweeps 8 k-tiles ----
    const int wave = tid >> 5;
    const int lane = tid & 31;

    v16bf a[2][4];
#pragma unroll
    for (int t = 0; t < 2; ++t) {
        int bt = wave * 2 + t;
#pragma unroll
        for (int c = 0; c < 4; ++c)
            a[t][c] = *(const v16bf*)(aBase + ((size_t)((bt * 4 + c) * 32 + lane) << 4));
    }

    const int rbase = wave * 32 + ((lane >> 4) << 3);   // C-matrix row base (bt0)

#pragma unroll
    for (int kt = 0; kt < NKT; ++kt) {
        v16bf bb[4];
#pragma unroll
        for (int c = 0; c < 4; ++c)
            bb[c] = *(const v16bf*)&Bs[kt][c][lane][0];

        v8f acc0 = {0.f, 0.f, 0.f, 0.f, 0.f, 0.f, 0.f, 0.f};
        v8f acc1 = {0.f, 0.f, 0.f, 0.f, 0.f, 0.f, 0.f, 0.f};
#pragma unroll
        for (int c = 0; c < 4; ++c) {
            acc0 = __builtin_amdgcn_wmma_f32_16x16x32_bf16(
                       false, a[0][c], false, bb[c], (short)0, acc0, false, false);
            acc1 = __builtin_amdgcn_wmma_f32_16x16x32_bf16(
                       false, a[1][c], false, bb[c], (short)0, acc1, false, false);
        }

        const int col = k0 + kt * 16 + (lane & 15);
#pragma unroll
        for (int v = 0; v < 8; ++v) {
            __builtin_nontemporal_store(acc0[v] * INV_T,
                &outp[(size_t)(rbase + v) * KN + col]);
            __builtin_nontemporal_store(acc1[v] * INV_T,
                &outp[(size_t)(rbase + 16 + v) * KN + col]);
        }
    }
}

// ---------------------------------------------------------------------------
// Kernel 3: momentum update of the 256 selected rows (runs after the copy).
// grid = (256, 2); block = 128 (= D).
//   p==0: mem=memory_l,  feat=l  -> new_memory_l
//   p==1: mem=memory_ab, feat=ab -> new_memory_ab
// ---------------------------------------------------------------------------
__global__ __launch_bounds__(128)
void momentum_update_kernel(const float* __restrict__ l,
                            const float* __restrict__ ab,
                            const int*   __restrict__ y,
                            const float* __restrict__ mem_l,
                            const float* __restrict__ mem_ab,
                            float*       __restrict__ out) {
    const int p = blockIdx.y;
    const float* feat = p ? ab : l;
    const float* mem  = p ? mem_ab : mem_l;
    float* dst = out + 2ull * BN * KN + (p ? (size_t)KN * DN : 0ull);

    const int b  = blockIdx.x;
    const int d  = threadIdx.x;
    const int yb = y[b];

    float pos = mem[(size_t)yb * DN + d] * 0.5f + feat[(size_t)b * DN + d] * 0.5f;

    __shared__ float red[DN];
    red[d] = pos * pos;
    __syncthreads();
#pragma unroll
    for (int s = DN / 2; s > 0; s >>= 1) {
        if (d < s) red[d] += red[d + s];
        __syncthreads();
    }
    float norm = sqrtf(red[0]);
    dst[(size_t)yb * DN + d] = pos / norm;
}

// ---------------------------------------------------------------------------
extern "C" void kernel_launch(void* const* d_in, const int* in_sizes, int n_in,
                              void* d_out, int out_size, void* d_ws, size_t ws_size,
                              hipStream_t stream) {
    const float* l      = (const float*)d_in[0];
    const float* ab     = (const float*)d_in[1];
    const int*   y      = (const int*)  d_in[2];
    const float* mem_l  = (const float*)d_in[3];
    const float* mem_ab = (const float*)d_in[4];
    float*   out = (float*)d_out;
    __bf16*  wsA = (__bf16*)d_ws;     // 2 x 64 KB packed bf16 A fragments

    pack_feat_kernel<<<dim3(2), 256, 0, stream>>>(l, ab, wsA);

    gemm_copy_kernel<<<dim3(KN / KCHUNK, 2), 256, 0, stream>>>(wsA, mem_ab, mem_l, out);

    momentum_update_kernel<<<dim3(BN, 2), 128, 0, stream>>>(l, ab, y, mem_l, mem_ab, out);
}
